// GNNUS_BaseModel_53326313947661
// MI455X (gfx1250) — matrix operand order repaired and checked by reading, not compiled, not run on hardware
//
#include <hip/hip_runtime.h>
#include <hip/hip_bf16.h>

// ---------------------------------------------------------------------------
// GNNUS_BaseModel for MI455X (gfx1250): compile-time-shaped WMMA f16->f32
// GEMMs + L2-resident float-atomic edge scatter.
// N=2048, B=64, F=48, E=65536, C=7, H=20.
// ---------------------------------------------------------------------------

typedef __attribute__((ext_vector_type(16))) _Float16 v16h;
typedef __attribute__((ext_vector_type(8)))  float    v8f;

#define NNODE 2048
#define BATCH 64
#define FEAT  48
#define NEDGE 65536
#define CLS   7
#define HID   20
#define MROWS (NNODE * BATCH)   // 131072 flattened (node,batch) rows; %16==0

static inline int cdiv(int a, int b) { return (a + b - 1) / b; }

// ---------------------------------------------------------------------------
// small utility kernels
// ---------------------------------------------------------------------------
__global__ void fill_zero_kernel(float* __restrict__ p, int n) {
  int i = blockIdx.x * blockDim.x + threadIdx.x;
  if (i < n) p[i] = 0.0f;
}

__global__ void cvt_f16_kernel(const float* __restrict__ src,
                               _Float16* __restrict__ dst, int n4) {
  int i = blockIdx.x * blockDim.x + threadIdx.x;   // 4 elements / thread
  if (i >= n4) return;
  const float4 v = ((const float4*)src)[i];
  _Float16 h0 = (_Float16)v.x, h1 = (_Float16)v.y;
  _Float16 h2 = (_Float16)v.z, h3 = (_Float16)v.w;
  short4 pk;
  pk.x = __builtin_bit_cast(short, h0);
  pk.y = __builtin_bit_cast(short, h1);
  pk.z = __builtin_bit_cast(short, h2);
  pk.w = __builtin_bit_cast(short, h3);
  ((short4*)dst)[i] = pk;
}

// deg[col[e]] += 1  (gcn_norm: degree over target index)
__global__ void degree_kernel(const int* __restrict__ col,
                              float* __restrict__ deg, int e) {
  int i = blockIdx.x * blockDim.x + threadIdx.x;
  if (i < e) atomicAdd(&deg[col[i]], 1.0f);
}

// norm[e] = dinv[row[e]] * dinv[col[e]],  dinv = deg>0 ? rsqrt(deg) : 0
__global__ void norm_kernel(const int* __restrict__ row,
                            const int* __restrict__ col,
                            const float* __restrict__ deg,
                            float* __restrict__ nrm, int e) {
  int i = blockIdx.x * blockDim.x + threadIdx.x;
  if (i >= e) return;
  float dr = deg[row[i]];
  float dc = deg[col[i]];
  float a = dr > 0.0f ? rsqrtf(dr) : 0.0f;
  float b = dc > 0.0f ? rsqrtf(dc) : 0.0f;
  nrm[i] = a * b;
}

// ---------------------------------------------------------------------------
// Compile-time-shaped WMMA GEMM:
//   C[M, Oa+Ob] = A[M,K](f16) * [Wa | Wb](f32, cvt to f16), K padded to 32/64.
//   Columns [0,Oa)       -> outa (f32 or f16) (+biasa if HAS_BA)
//   Columns [Oa, Oa+Ob)  -> outb (f32)        (+biasb if HAS_BB)
// One wave per 16-row M tile, grid-stride over M/16 tiles. Weight panel and
// bias values are register-resident for the whole kernel. All tile counts,
// K-padding predicates and address strides are compile-time constants so the
// inner loop is a straight-line block of v_wmma_f32_16x16x32_f16, and the
// store epilogue uses one base address per column tile with immediate
// row offsets (v*O is a compile-time constant).
// Requires M % 16 == 0.
// ---------------------------------------------------------------------------
template <int K, int Oa, int Ob, bool AF16OUT, bool HAS_BA, bool HAS_BB>
__global__ __launch_bounds__(256)
void wmma_gemm_t(const _Float16* __restrict__ A, int M,
                 const float* __restrict__ Wa,
                 const float* __restrict__ biasa,
                 float* __restrict__ outa_f32,
                 _Float16* __restrict__ outa_f16,
                 const float* __restrict__ Wb,
                 const float* __restrict__ biasb,
                 float* __restrict__ outb_f32) {
  constexpr int Ot = Oa + Ob;
  constexpr int NT = (Ot + 15) / 16;   // column tiles
  constexpr int KT = (K + 31) / 32;    // k tiles
  constexpr int LDA = K;               // A is packed [M,K]

  const int lane = threadIdx.x & 31;
  const int n16  = lane & 15;          // column (B/C/D) or row (A) within tile
  const int hi   = lane >> 4;          // lane-half selector
  const int wavesPerBlk = blockDim.x >> 5;
  const int wave  = blockIdx.x * wavesPerBlk + (threadIdx.x >> 5);
  const int nwave = gridDim.x * wavesPerBlk;

  // ---- weight panel -> registers (B tile, 32x16 f16, contiguous k order; lane holds
  //      column N=lane&15, half e -> K = e + 16*(lane>=16)) ----
  v16h Bt[KT][NT];
#pragma unroll
  for (int kt = 0; kt < KT; ++kt) {
#pragma unroll
    for (int nt = 0; nt < NT; ++nt) {
      v16h b;
#pragma unroll
      for (int e = 0; e < 16; ++e) {
        const int k   = kt * 32 + e + hi * 16;
        const int col = nt * 16 + n16;
        float w = 0.0f;
        if (k < K && col < Ot)
          w = (col < Oa) ? Wa[k * Oa + col] : Wb[k * Ob + (col - Oa)];
        b[e] = (_Float16)w;
      }
      Bt[kt][nt] = b;
    }
  }

  // ---- per-lane bias values (output column is loop-invariant) ----
  float bias[NT];
#pragma unroll
  for (int nt = 0; nt < NT; ++nt) {
    const int col = nt * 16 + n16;
    float bv = 0.0f;
    if (HAS_BA && col < Oa) bv = biasa[col];
    if (HAS_BB && col >= Oa && col < Ot) bv = biasb[col - Oa];
    bias[nt] = bv;
  }

  const int tiles = M >> 4;            // M % 16 == 0
  for (int t = wave; t < tiles; t += nwave) {
    const int m0 = t << 4;

    // ---- A tiles (16x32 f16: row M = lane&15;
    //      e -> K = kt*32 + e + 16*hi, matching B)
    const _Float16* ap = A + (size_t)(m0 + n16) * LDA;
    v16h Av[KT];
#pragma unroll
    for (int kt = 0; kt < KT; ++kt) {
      v16h a;
#pragma unroll
      for (int e = 0; e < 16; ++e) {
        const int k = kt * 32 + e + hi * 16;
        a[e] = (k < K) ? ap[k] : (_Float16)0.0f;
      }
      Av[kt] = a;
    }

    // ---- straight-line WMMA block ----
    v8f acc[NT] = {};
#pragma unroll
    for (int kt = 0; kt < KT; ++kt) {
#pragma unroll
      for (int nt = 0; nt < NT; ++nt) {
        acc[nt] = __builtin_amdgcn_wmma_f32_16x16x32_f16(
            false, Av[kt], false, Bt[kt][nt], (short)0, acc[nt],
            false, false);
      }
    }

    // ---- store epilogue (C/D layout: N = lane&15, VGPR v -> M = v + 8*hi).
    //      One base address per column tile; row offsets v*O are immediates.
    const size_t rbase = (size_t)(m0 + hi * 8);
#pragma unroll
    for (int nt = 0; nt < NT; ++nt) {
      const int col = nt * 16 + n16;
      if (col < Oa) {
        if constexpr (AF16OUT) {
          _Float16* po = outa_f16 + rbase * Oa + col;
#pragma unroll
          for (int v = 0; v < 8; ++v) {
            float val = acc[nt][v];
            if constexpr (HAS_BA) val += bias[nt];
            po[v * Oa] = (_Float16)val;
          }
        } else {
          float* po = outa_f32 + rbase * Oa + col;
#pragma unroll
          for (int v = 0; v < 8; ++v) {
            float val = acc[nt][v];
            if constexpr (HAS_BA) val += bias[nt];
            po[v * Oa] = val;
          }
        }
      } else {
        if constexpr (Ob > 0) {
          if (col < Ot) {
            float* po = outb_f32 + rbase * Ob + (col - Oa);
#pragma unroll
            for (int v = 0; v < 8; ++v) {
              float val = acc[nt][v];
              if constexpr (HAS_BB) val += bias[nt];
              po[v * Ob] = val;
            }
          }
        }
      }
    }
  }
}

// ---------------------------------------------------------------------------
// edge scatter: buf[col[e], :, :] += h[row[e], :, :] * norm[e]
// features in [N, B, O] layout -> one contiguous BO-float block per node;
// whole working set (~10 MB) lives in the 192 MB L2, float atomics RMW there.
// ---------------------------------------------------------------------------
__global__ void scatter_kernel(const float* __restrict__ h,
                               float* __restrict__ buf,
                               const int* __restrict__ row,
                               const int* __restrict__ col,
                               const float* __restrict__ nrm,
                               int E, int BO) {
  const int per   = BO >> 2;            // BO divisible by 4 (1280 / 448)
  const int total = E * per;
  int gid = blockIdx.x * blockDim.x + threadIdx.x;
  if (gid >= total) return;
  const int e = gid / per;
  const int j = (gid - e * per) << 2;
  const float s = nrm[e];
  const float* hp = h   + (size_t)row[e] * BO + j;
  float*       bp = buf + (size_t)col[e] * BO + j;
#pragma unroll
  for (int t = 0; t < 4; ++t) atomicAdd(&bp[t], hp[t] * s);
}

// y = elu(gelu(z)); write f16 for the next WMMA GEMM
__global__ void act1_kernel(const float* __restrict__ in,
                            _Float16* __restrict__ out, int n) {
  int i = blockIdx.x * blockDim.x + threadIdx.x;
  if (i >= n) return;
  float z = in[i];
  float g = 0.5f * z * (1.0f + erff(z * 0.70710678118654752f));   // exact gelu
  float y = (g > 0.0f) ? g : (expf(g) - 1.0f);                    // elu
  out[i] = (_Float16)y;
}

// softmax over batch axis (legacy dim=0 of [B,N,C]); optional relu first,
// optional accumulate into destination.
__global__ void softmax_b_kernel(const float* __restrict__ in,
                                 float* __restrict__ out,
                                 int nNodes, int Bb, int Cc,
                                 int do_relu, int do_acc) {
  int gid = blockIdx.x * blockDim.x + threadIdx.x;
  if (gid >= nNodes * Cc) return;
  const int n = gid / Cc;
  const int c = gid - n * Cc;
  const float* p = in + (size_t)n * Bb * Cc + c;
  float mx = -3.4e38f;
  for (int b = 0; b < Bb; ++b) {
    float v = p[b * Cc];
    if (do_relu) v = fmaxf(v, 0.0f);
    mx = fmaxf(mx, v);
  }
  float sum = 0.0f;
  for (int b = 0; b < Bb; ++b) {
    float v = p[b * Cc];
    if (do_relu) v = fmaxf(v, 0.0f);
    sum += expf(v - mx);
  }
  const float inv = 1.0f / sum;
  float* q = out + (size_t)n * Bb * Cc + c;
  for (int b = 0; b < Bb; ++b) {
    float v = p[b * Cc];
    if (do_relu) v = fmaxf(v, 0.0f);
    float r = expf(v - mx) * inv;
    if (do_acc) q[b * Cc] += r; else q[b * Cc] = r;
  }
}

// tiny 7x7 head: out[m,c] = sum_k scale*(in0[m,k]+in1[m,k]) * W[k,c] + bias[c]
__global__ void head_kernel(const float* __restrict__ in0,
                            const float* __restrict__ in1, float scale,
                            const float* __restrict__ W,
                            const float* __restrict__ bias,
                            float* __restrict__ out, int M, int C) {
  int gid = blockIdx.x * blockDim.x + threadIdx.x;
  if (gid >= M * C) return;
  const int m = gid / C;
  const int c = gid - m * C;
  const float* a = in0 + (size_t)m * C;
  const float* b = in1 ? (in1 + (size_t)m * C) : nullptr;
  float s = bias[c];
  for (int k = 0; k < C; ++k) {
    float v = a[k] + (b ? b[k] : 0.0f);
    s += scale * v * W[k * C + c];
  }
  out[gid] = s;
}

// [ (n,b), C ] -> [N, C, B]
__global__ void out_transpose_kernel(const float* __restrict__ in,
                                     float* __restrict__ out,
                                     int nNodes, int Bb, int Cc) {
  int gid = blockIdx.x * blockDim.x + threadIdx.x;
  if (gid >= nNodes * Bb * Cc) return;
  const int b   = gid % Bb;
  const int tmp = gid / Bb;
  const int c   = tmp % Cc;
  const int n   = tmp / Cc;
  out[gid] = in[((size_t)n * Bb + b) * Cc + c];
}

// ---------------------------------------------------------------------------
// host driver
// ---------------------------------------------------------------------------
extern "C" void kernel_launch(void* const* d_in, const int* in_sizes, int n_in,
                              void* d_out, int out_size, void* d_ws,
                              size_t ws_size, hipStream_t stream) {
  (void)in_sizes; (void)n_in; (void)out_size; (void)ws_size;

  // ---- inputs (setup_inputs order) ----
  const int* edgesA = (const int*)d_in[0];
  const int* edgesW = (const int*)d_in[1];
  const int* edgesE = (const int*)d_in[2];
  const int* edgesL = (const int*)d_in[3];
  const float* feat[6] = {
    (const float*)d_in[4], (const float*)d_in[5], (const float*)d_in[6],
    (const float*)d_in[7], (const float*)d_in[8], (const float*)d_in[9]
  };
  const float* W1  = (const float*)d_in[10];
  const float* V1  = (const float*)d_in[11];
  const float* b1  = (const float*)d_in[12];
  const float* W2  = (const float*)d_in[13];
  const float* V2  = (const float*)d_in[14];
  const float* b2  = (const float*)d_in[15];
  const float* Wlt = (const float*)d_in[16];
  const float* blt = (const float*)d_in[17];
  const float* Wl2 = (const float*)d_in[18];
  const float* bl2 = (const float*)d_in[19];
  const float* Wd  = (const float*)d_in[20];
  const float* bd  = (const float*)d_in[21];
  const float* Wg  = (const float*)d_in[22];
  const float* bg  = (const float*)d_in[23];
  float* out = (float*)d_out;

  const int M = MROWS;

  // ---- workspace bump allocator ----
  char* ws = (char*)d_ws;
  size_t off = 0;
  auto alloc = [&](size_t bytes) -> void* {
    void* p = ws + off;
    off += (bytes + 255) & ~(size_t)255;
    return p;
  };
  float*    norm4[4];
  for (int a = 0; a < 4; ++a) norm4[a] = (float*)alloc((size_t)NEDGE * 4);
  float*    deg  = (float*)   alloc((size_t)NNODE * 4);
  _Float16* x16  = (_Float16*)alloc((size_t)M * FEAT * 2);
  float*    h1   = (float*)   alloc((size_t)M * HID * 4);
  float*    buf1 = (float*)   alloc((size_t)M * HID * 4);
  _Float16* y16  = (_Float16*)alloc((size_t)M * HID * 2);
  float*    h2   = (float*)   alloc((size_t)M * CLS * 4);
  float*    buf2 = (float*)   alloc((size_t)M * CLS * 4);
  float*    o_l  = (float*)   alloc((size_t)M * CLS * 4);
  float*    gnn  = (float*)   alloc((size_t)M * CLS * 4);
  float*    o_lt = (float*)   alloc((size_t)M * CLS * 4);
  float*    dpre = (float*)   alloc((size_t)M * CLS * 4);
  float*    gpre = (float*)   alloc((size_t)M * CLS * 4);
  float*    tmpD = (float*)   alloc((size_t)M * CLS * 4);
  _Float16* lt16 = (_Float16*)alloc((size_t)M * 40 * 2);

  const int TPB = 256;

  // ---- gcn norms for the 4 graphs ----
  const int* graphs[4] = { edgesA, edgesW, edgesE, edgesL };
  for (int a = 0; a < 4; ++a) {
    const int* row = graphs[a];
    const int* col = graphs[a] + NEDGE;
    fill_zero_kernel<<<cdiv(NNODE, TPB), TPB, 0, stream>>>(deg, NNODE);
    degree_kernel<<<cdiv(NEDGE, TPB), TPB, 0, stream>>>(col, deg, NEDGE);
    norm_kernel<<<cdiv(NEDGE, TPB), TPB, 0, stream>>>(row, col, deg,
                                                      norm4[a], NEDGE);
  }

  // ---- accumulator for the 5 summed GNN branches ----
  fill_zero_kernel<<<cdiv(M * CLS, TPB), TPB, 0, stream>>>(gnn, M * CLS);

  const int gemmBlocks = 1024;  // 8 waves/block -> 8192 waves = #M-tiles
  const int branchGraph[6] = { 0, 1, 2, 0, 0, 3 };

  for (int i = 0; i < 6; ++i) {
    const int* row  = graphs[branchGraph[i]];
    const int* col  = graphs[branchGraph[i]] + NEDGE;
    const float* nr = norm4[branchGraph[i]];

    // x -> f16 (float4 -> 4x f16 per thread)
    cvt_f16_kernel<<<cdiv(M * FEAT / 4, TPB), TPB, 0, stream>>>(
        feat[i], x16, M * FEAT / 4);

    // layer 1: h1 = x@W1[i] ; buf1 = x@V1[i] + b1[i]
    wmma_gemm_t<FEAT, HID, HID, false, false, true>
        <<<gemmBlocks, TPB, 0, stream>>>(
            x16, M,
            W1 + (size_t)i * FEAT * HID, nullptr, h1, nullptr,
            V1 + (size_t)i * FEAT * HID, b1 + (size_t)i * HID, buf1);

    // buf1 += A_hat @ h1
    {
      int total = NEDGE * ((BATCH * HID) >> 2);
      scatter_kernel<<<cdiv(total, TPB), TPB, 0, stream>>>(
          h1, buf1, row, col, nr, NEDGE, BATCH * HID);
    }
    // y = elu(gelu(buf1)) -> f16
    act1_kernel<<<cdiv(M * HID, TPB), TPB, 0, stream>>>(buf1, y16, M * HID);

    // layer 2: h2 = y@W2[i] ; buf2 = y@V2[i] + b2[i]
    wmma_gemm_t<HID, CLS, CLS, false, false, true>
        <<<gemmBlocks, TPB, 0, stream>>>(
            y16, M,
            W2 + (size_t)i * HID * CLS, nullptr, h2, nullptr,
            V2 + (size_t)i * HID * CLS, b2 + (size_t)i * CLS, buf2);
    {
      int total = NEDGE * ((BATCH * CLS) >> 2);
      scatter_kernel<<<cdiv(total, TPB), TPB, 0, stream>>>(
          h2, buf2, row, col, nr, NEDGE, BATCH * CLS);
    }
    // softmax_b(relu(buf2)); branches 0..4 accumulate, branch 5 stores o_l
    softmax_b_kernel<<<cdiv(NNODE * CLS, TPB), TPB, 0, stream>>>(
        buf2, (i < 5) ? gnn : o_l, NNODE, BATCH, CLS,
        /*relu=*/1, /*acc=*/(i < 5) ? 1 : 0);
  }

  // ---- dense path (x16 still holds Location_time in f16) ----
  // lt16 = x@Wlt + blt (f16 out for the next WMMA stage)
  wmma_gemm_t<FEAT, 40, 0, true, true, false>
      <<<gemmBlocks, TPB, 0, stream>>>(
          x16, M, Wlt, blt, nullptr, lt16, nullptr, nullptr, nullptr);
  // buf2 = lt16@Wl2 + bl2
  wmma_gemm_t<40, CLS, 0, false, true, false>
      <<<gemmBlocks, TPB, 0, stream>>>(
          lt16, M, Wl2, bl2, buf2, nullptr, nullptr, nullptr, nullptr);
  softmax_b_kernel<<<cdiv(NNODE * CLS, TPB), TPB, 0, stream>>>(
      buf2, o_lt, NNODE, BATCH, CLS, /*relu=*/0, /*acc=*/0);

  // ---- heads ----
  // dpre = (2*o_l + 2*o_lt) @ Wd + bd
  head_kernel<<<cdiv(M * CLS, TPB), TPB, 0, stream>>>(
      o_l, o_lt, 2.0f, Wd, bd, dpre, M, CLS);
  // gpre = gnn @ Wg + bg
  head_kernel<<<cdiv(M * CLS, TPB), TPB, 0, stream>>>(
      gnn, nullptr, 1.0f, Wg, bg, gpre, M, CLS);

  // tmpD = sm0(dpre) + sm0(gpre)
  softmax_b_kernel<<<cdiv(NNODE * CLS, TPB), TPB, 0, stream>>>(
      dpre, tmpD, NNODE, BATCH, CLS, 0, /*acc=*/0);
  softmax_b_kernel<<<cdiv(NNODE * CLS, TPB), TPB, 0, stream>>>(
      gpre, tmpD, NNODE, BATCH, CLS, 0, /*acc=*/1);

  // [N*B, C] -> [N, C, B]
  out_transpose_kernel<<<cdiv(NNODE * BATCH * CLS, TPB), TPB, 0, stream>>>(
      tmpD, out, NNODE, BATCH, CLS);
}
